// Attention_49074296324413
// MI455X (gfx1250) — matrix operand order repaired and checked by reading, not compiled, not run on hardware
//
#include <hip/hip_runtime.h>

typedef __bf16 bf16_t;
typedef bf16_t v16bf __attribute__((ext_vector_type(16)));
typedef float  v8f   __attribute__((ext_vector_type(8)));

union Frag16 { unsigned short u[16]; v16bf v; };

// ---------------------------------------------------------------------------
// Generic batched GEMM: C[M,N] = A[M,K] * B[K,N], bf16 in, f32 accumulate.
// BTRANS: B element (k,n) fetched from B[n*ldb + k] (B given as N x K).
// OUT_BF16: store C as bf16 instead of f32.
// Block = 128 threads (4 waves). Tile 64x64, BK=64 (two wmma K-steps/stage),
// double-buffered LDS with explicit load->compute->store pipelining.
// Wave w computes rows [16w,16w+16) x all 64 cols -> 4 accumulators.
// Requires lda%4==0, ldb%4==0, K%16==0, N%16==0, 8-byte aligned A/B bases
// (all call sites satisfy this).
// ---------------------------------------------------------------------------
template<bool BTRANS, bool OUT_BF16>
__global__ __launch_bounds__(128)
void gemm_bf16_wmma(const bf16_t* __restrict__ A, const bf16_t* __restrict__ Bm,
                    void* __restrict__ Cm,
                    int M, int N, int K, int lda, int ldb, int ldc,
                    long long sA, long long sB, long long sC)
{
    __shared__ unsigned short As[2][64][68];   // [buf][m][k], row 136B (8B aligned)
    __shared__ unsigned short Bs[2][64][68];   // [buf][n][k] (transposed stage)

    const int tid  = threadIdx.x;
    const int lane = tid & 31;
    const int wv   = tid >> 5;
    const int bn   = blockIdx.x * 64;
    const int bm   = blockIdx.y * 64;
    const int bz   = blockIdx.z;

    const bf16_t* Ab = A  + (size_t)bz * (size_t)sA;
    const bf16_t* Bb = Bm + (size_t)bz * (size_t)sB;

    const int r0  = tid >> 4;        // base row for staging (stride 8 over p)
    const int s4  = (tid & 15) * 4;  // 4-element segment within a 64-wide row

    v8f acc[4];
    #pragma unroll
    for (int t = 0; t < 4; ++t)
        #pragma unroll
        for (int r = 0; r < 8; ++r) acc[t][r] = 0.0f;

    uint2 av[8], bv[8];              // in-flight tile (A rows / B rows)

    // ---- issue global loads for tile at k0 (branchless, zero-filled) ----
    auto loadTile = [&](int k0) {
        #pragma unroll
        for (int p = 0; p < 8; ++p) {
            int rr = r0 + p * 8;
            {   // A: row gm, k segment [k0+s4, +4)
                int gm = bm + rr;
                bool ok = (gm < M) && (k0 + s4 + 4 <= K);
                size_t o = ok ? ((size_t)gm * lda + (k0 + s4)) : 0;
                uint2 v = *(const uint2*)(Ab + o);
                av[p].x = ok ? v.x : 0u;
                av[p].y = ok ? v.y : 0u;
            }
            if (BTRANS) {   // B': row n, k segment (contiguous k in memory)
                int gn = bn + rr;
                bool ok = (gn < N) && (k0 + s4 + 4 <= K);
                size_t o = ok ? ((size_t)gn * ldb + (k0 + s4)) : 0;
                uint2 v = *(const uint2*)(Bb + o);
                bv[p].x = ok ? v.x : 0u;
                bv[p].y = ok ? v.y : 0u;
            } else {        // B: row k, n segment (contiguous n in memory)
                int gk = k0 + rr;
                bool ok = (gk < K) && (bn + s4 + 4 <= N);
                size_t o = ok ? ((size_t)gk * ldb + (bn + s4)) : 0;
                uint2 v = *(const uint2*)(Bb + o);
                bv[p].x = ok ? v.x : 0u;
                bv[p].y = ok ? v.y : 0u;
            }
        }
        // prefetch one more tile ahead (global_prefetch_b8)
        if ((bm + 64 <= M) && (bn + 64 <= N) && (k0 + 128 <= K)) {
            __builtin_prefetch(Ab + (size_t)(bm + (tid >> 1)) * lda + k0 + 64 + (tid & 1) * 32, 0, 0);
            if (!BTRANS)
                __builtin_prefetch(Bb + (size_t)(k0 + 64 + (tid >> 1)) * ldb + bn + (tid & 1) * 32, 0, 0);
            else
                __builtin_prefetch(Bb + (size_t)(bn + (tid >> 1)) * ldb + k0 + 64 + (tid & 1) * 32, 0, 0);
        }
    };

    // ---- drain in-flight tile into LDS buffer ----
    auto storeTile = [&](int bufi) {
        #pragma unroll
        for (int p = 0; p < 8; ++p) {
            int rr = r0 + p * 8;
            *(uint2*)&As[bufi][rr][s4] = av[p];
            if (BTRANS) {
                *(uint2*)&Bs[bufi][rr][s4] = bv[p];
            } else {
                const unsigned short* pv = (const unsigned short*)&bv[p];
                Bs[bufi][s4 + 0][rr] = pv[0];
                Bs[bufi][s4 + 1][rr] = pv[1];
                Bs[bufi][s4 + 2][rr] = pv[2];
                Bs[bufi][s4 + 3][rr] = pv[3];
            }
        }
    };

    // ---- 8 WMMAs over one 64-deep LDS tile ----
    auto compute = [&](int bufi) {
        #pragma unroll
        for (int ks = 0; ks < 64; ks += 32) {
            Frag16 af;
            {
                int ml  = wv * 16 + (lane & 15);
                int kbA = ks + ((lane < 16) ? 0 : 8);
                #pragma unroll
                for (int h = 0; h < 16; ++h)
                    af.u[h] = As[bufi][ml][kbA + (h & 7) + ((h >> 3) << 4)];
            }
            int kbB = ks + ((lane < 16) ? 0 : 8);
            #pragma unroll
            for (int t = 0; t < 4; ++t) {
                Frag16 bfv;
                int nl = t * 16 + (lane & 15);
                #pragma unroll
                for (int h = 0; h < 16; ++h) bfv.u[h] = Bs[bufi][nl][kbB + (h & 7) + ((h >> 3) << 4)];
                acc[t] = __builtin_amdgcn_wmma_f32_16x16x32_bf16(
                    false, af.v, false, bfv.v, (short)0, acc[t], false, false);
            }
        }
    };

    // ---- pipelined main loop: load(k+1) | compute(k) | drain(k+1) | barrier ----
    loadTile(0);
    storeTile(0);
    __syncthreads();
    int buf = 0;
    for (int k0 = 0; k0 < K; k0 += 64) {
        bool more = (k0 + 64) < K;
        if (more) loadTile(k0 + 64);
        compute(buf);
        if (more) storeTile(buf ^ 1);
        __syncthreads();
        buf ^= 1;
    }

    // ---- store C per 16x16 f32 C/D layout ----
    const int mo = ((lane >> 4) << 3);
    #pragma unroll
    for (int t = 0; t < 4; ++t) {
        int gn = bn + t * 16 + (lane & 15);
        if (gn >= N) continue;
        #pragma unroll
        for (int r = 0; r < 8; ++r) {
            int gm = bm + wv * 16 + r + mo;
            if (gm < M) {
                size_t o = (size_t)bz * (size_t)sC + (size_t)gm * ldc + gn;
                if (OUT_BF16) ((bf16_t*)Cm)[o] = (bf16_t)acc[t][r];
                else          ((float*)Cm)[o]  = acc[t][r];
            }
        }
    }
}

// ---------------------------------------------------------------------------
// Vectorized f32 -> bf16 conversion: 4 elements per thread (n % 4 == 0).
__global__ void f32_to_bf16_v4(const float* __restrict__ in, bf16_t* __restrict__ out,
                               long long n4)
{
    long long i = (long long)blockIdx.x * blockDim.x + threadIdx.x;
    if (i >= n4) return;
    float4 v = ((const float4*)in)[i];
    bf16_t r[4] = { (bf16_t)v.x, (bf16_t)v.y, (bf16_t)v.z, (bf16_t)v.w };
    ((uint2*)out)[i] = *(const uint2*)r;
}

// Depthwise 3x3 (pad=1) on kv [8,768,64,64] (bf16). Channels <384 -> k path:
// fused 2x2 mean pool, f32 out [8,384,32,32]. Channels >=384 -> v path: bf16
// out [8,384,64,64]. One block per (b,ch) plane.
__global__ __launch_bounds__(256)
void dwconv3x3(const bf16_t* __restrict__ kvb, const float* __restrict__ wdw,
               float* __restrict__ kp, bf16_t* __restrict__ vb)
{
    int bc = blockIdx.x;          // b*768 + ch
    int b  = bc / 768, ch = bc % 768;
    const bf16_t* in = kvb + (size_t)bc * 4096;
    float w[9];
    #pragma unroll
    for (int i = 0; i < 9; ++i) w[i] = wdw[ch * 9 + i];

    auto dw = [&](int y, int x) -> float {
        float s = 0.0f;
        #pragma unroll
        for (int ky = 0; ky < 3; ++ky) {
            int sy = y + ky - 1;
            if ((unsigned)sy >= 64u) continue;
            #pragma unroll
            for (int kx = 0; kx < 3; ++kx) {
                int sx = x + kx - 1;
                if ((unsigned)sx >= 64u) continue;
                s += w[ky * 3 + kx] * (float)in[sy * 64 + sx];
            }
        }
        return s;
    };

    if (ch < 384) {
        float* out = kp + ((size_t)b * 384 + ch) * 1024;
        for (int p = threadIdx.x; p < 1024; p += 256) {
            int py = p >> 5, px = p & 31;
            float s = dw(2*py, 2*px) + dw(2*py, 2*px+1)
                    + dw(2*py+1, 2*px) + dw(2*py+1, 2*px+1);
            out[p] = 0.25f * s;
        }
    } else {
        bf16_t* out = vb + ((size_t)b * 384 + (ch - 384)) * 4096;
        for (int p = threadIdx.x; p < 4096; p += 256)
            out[p] = (bf16_t)dw(p >> 6, p & 63);
    }
}

// im2col for the FULL 3x3 conv (pad=1) on q1 [8,384,32,32] bf16 ->
// col [8, 3456, 1024] bf16, K ordered as ci*9 + ky*3 + kx (matches w_q_dw flat).
__global__ void im2col3x3(const bf16_t* __restrict__ q1, bf16_t* __restrict__ col,
                          long long total)
{
    long long idx = (long long)blockIdx.x * blockDim.x + threadIdx.x;
    if (idx >= total) return;
    int n = (int)(idx & 1023);
    long long t = idx >> 10;
    int k = (int)(t % 3456);
    int b = (int)(t / 3456);
    int kx = k % 3, ky = (k / 3) % 3, ci = k / 9;
    int y = n >> 5, x = n & 31;
    int sy = y + ky - 1, sx = x + kx - 1;
    bool ok = ((unsigned)sy < 32u) && ((unsigned)sx < 32u);
    size_t src = (((size_t)b * 384 + ci) << 10) + (size_t)(ok ? (sy << 5) + sx : 0);
    bf16_t v = q1[src];
    col[idx] = ok ? v : (bf16_t)0.0f;
}

// L2-normalize rows of length L (f32 in -> bf16 out). One block per row.
__global__ __launch_bounds__(256)
void l2norm_rows(const float* __restrict__ in, bf16_t* __restrict__ out, int L)
{
    __shared__ float red[256];
    size_t row = blockIdx.x;
    const float* r = in + row * (size_t)L;
    float s = 0.0f;
    for (int i = threadIdx.x; i < L; i += 256) { float v = r[i]; s += v * v; }
    red[threadIdx.x] = s;
    __syncthreads();
    for (int st = 128; st > 0; st >>= 1) {
        if ((int)threadIdx.x < st) red[threadIdx.x] += red[threadIdx.x + st];
        __syncthreads();
    }
    float inv = 1.0f / fmaxf(sqrtf(red[0]), 1e-12f);
    bf16_t* o = out + row * (size_t)L;
    for (int i = threadIdx.x; i < L; i += 256) o[i] = (bf16_t)(r[i] * inv);
}

// softmax over rows of 48 (with per-head temperature), f32 in -> bf16 out.
// row = (b*8 + h)*48 + c.
__global__ __launch_bounds__(64)
void softmax48(const float* __restrict__ attn, const float* __restrict__ temp,
               bf16_t* __restrict__ out)
{
    __shared__ float red[64];
    int row = blockIdx.x;
    int h   = (row / 48) & 7;
    const float* r = attn + (size_t)row * 48;
    float t = temp[h];
    int i = threadIdx.x;
    float v = (i < 48) ? r[i] * t : -1e30f;
    red[i] = v; __syncthreads();
    for (int st = 32; st > 0; st >>= 1) {
        if (i < st) red[i] = fmaxf(red[i], red[i + st]);
        __syncthreads();
    }
    float m = red[0]; __syncthreads();
    float e = (i < 48) ? __expf(v - m) : 0.0f;
    red[i] = e; __syncthreads();
    for (int st = 32; st > 0; st >>= 1) {
        if (i < st) red[i] += red[i + st];
        __syncthreads();
    }
    float inv = 1.0f / red[0];
    if (i < 48) out[(size_t)row * 48 + i] = (bf16_t)(e * inv);
}

// ---------------------------------------------------------------------------
extern "C" void kernel_launch(void* const* d_in, const int* in_sizes, int n_in,
                              void* d_out, int out_size, void* d_ws, size_t ws_size,
                              hipStream_t stream)
{
    const float* x      = (const float*)d_in[0];  // [8,384,64,64]
    const float* y      = (const float*)d_in[1];  // [8,384,32,32]
    const float* w_kv   = (const float*)d_in[2];  // [768,384]
    const float* w_kvdw = (const float*)d_in[3];  // [768,1,3,3]
    const float* w_q    = (const float*)d_in[4];  // [384,384]
    const float* w_qdw  = (const float*)d_in[5];  // [384,384,3,3]
    const float* w_proj = (const float*)d_in[6];  // [384,384]
    const float* temp   = (const float*)d_in[7];  // [8]

    const int Bn = 8, Cc = 384, C2 = 768, N1 = 4096, N2 = 1024, K9 = 3456;

    size_t off = 0;
    auto alloc = [&](size_t bytes) {
        size_t o = off; off += (bytes + 255) & ~(size_t)255; return o;
    };
    char* ws = (char*)d_ws;

    bf16_t* xb     = (bf16_t*)(ws + alloc((size_t)Bn*Cc*N1 * 2));  // 25.2 MB
    bf16_t* yb     = (bf16_t*)(ws + alloc((size_t)Bn*Cc*N2 * 2));  //  6.3 MB
    bf16_t* wkvb   = (bf16_t*)(ws + alloc((size_t)C2*Cc * 2));
    bf16_t* wqb    = (bf16_t*)(ws + alloc((size_t)Cc*Cc * 2));
    bf16_t* wqdwb  = (bf16_t*)(ws + alloc((size_t)Cc*K9 * 2));
    bf16_t* wprojb = (bf16_t*)(ws + alloc((size_t)Cc*Cc * 2));
    bf16_t* kvb    = (bf16_t*)(ws + alloc((size_t)Bn*C2*N1 * 2)); // 50.3 MB
    float*  kp     = (float* )(ws + alloc((size_t)Bn*Cc*N2 * 4)); // 12.6 MB pooled k
    bf16_t* vb     = (bf16_t*)(ws + alloc((size_t)Bn*Cc*N1 * 2)); // 25.2 MB
    bf16_t* q1b    = (bf16_t*)(ws + alloc((size_t)Bn*Cc*N2 * 2));
    bf16_t* colb   = (bf16_t*)(ws + alloc((size_t)Bn*K9*N2 * 2)); // 56.6 MB (L2-resident)
    float*  qf     = (float* )(ws + alloc((size_t)Bn*Cc*N2 * 4));
    bf16_t* qb     = (bf16_t*)(ws + alloc((size_t)Bn*Cc*N2 * 2));
    bf16_t* kb     = (bf16_t*)(ws + alloc((size_t)Bn*Cc*N2 * 2));
    float*  attn_s = (float* )(ws + alloc((size_t)Bn*8*48*48 * 4));
    bf16_t* attn_b = (bf16_t*)(ws + alloc((size_t)Bn*8*48*48 * 2));
    bf16_t* ob     = (bf16_t*)(ws + alloc((size_t)Bn*Cc*N1 * 2)); // 25.2 MB

    auto cvt = [&](const float* src, bf16_t* dst, long long n) {
        long long n4 = n >> 2;
        f32_to_bf16_v4<<<dim3((unsigned)((n4 + 255) / 256)), dim3(256), 0, stream>>>(src, dst, n4);
    };

    // 1. convert activations + weights to bf16
    cvt(x,      xb,     (long long)Bn*Cc*N1);
    cvt(y,      yb,     (long long)Bn*Cc*N2);
    cvt(w_kv,   wkvb,   (long long)C2*Cc);
    cvt(w_q,    wqb,    (long long)Cc*Cc);
    cvt(w_qdw,  wqdwb,  (long long)Cc*K9);
    cvt(w_proj, wprojb, (long long)Cc*Cc);

    // 2. kv = Wkv(768x384) x X(384x4096), batch 8 -> bf16
    gemm_bf16_wmma<false, true><<<dim3(N1/64, C2/64, Bn), dim3(128), 0, stream>>>(
        wkvb, xb, kvb, C2, N1, Cc, Cc, N1, N1, 0, (long long)Cc*N1, (long long)C2*N1);

    // 3. depthwise 3x3 on kv; k-path fused 2x2 mean pool (f32), v-path bf16
    dwconv3x3<<<dim3(Bn*C2), dim3(256), 0, stream>>>(kvb, w_kvdw, kp, vb);

    // 4. q1 = Wq(384x384) x Y(384x1024), batch 8 -> bf16
    gemm_bf16_wmma<false, true><<<dim3(N2/64, Cc/64, Bn), dim3(128), 0, stream>>>(
        wqb, yb, q1b, Cc, N2, Cc, Cc, N2, N2, 0, (long long)Cc*N2, (long long)Cc*N2);

    // 5. im2col of q1 for the full 3x3 conv
    {
        long long tot = (long long)Bn*K9*N2;
        im2col3x3<<<dim3((unsigned)((tot + 255) / 256)), dim3(256), 0, stream>>>(q1b, colb, tot);
    }

    // 6. q = Wqdw(384x3456) x col(3456x1024), batch 8 -> f32
    gemm_bf16_wmma<false, false><<<dim3(N2/64, Cc/64, Bn), dim3(128), 0, stream>>>(
        wqdwb, colb, qf, Cc, N2, K9, K9, N2, N2, 0, (long long)K9*N2, (long long)Cc*N2);

    // 7. L2-normalize q and pooled k along the spatial axis (1024) -> bf16
    l2norm_rows<<<dim3(Bn*Cc), dim3(256), 0, stream>>>(qf, qb, N2);
    l2norm_rows<<<dim3(Bn*Cc), dim3(256), 0, stream>>>(kp, kb, N2);

    // 8. attn = q(48x1024) x k^T(1024x48), batch 64 (b*h) -> f32
    gemm_bf16_wmma<true, false><<<dim3(1, 1, Bn*8), dim3(128), 0, stream>>>(
        qb, kb, attn_s, 48, 48, N2, N2, N2, 48,
        (long long)48*N2, (long long)48*N2, (long long)48*48);

    // 9. softmax(attn * temperature[h]) -> bf16
    softmax48<<<dim3(Bn*8*48), dim3(64), 0, stream>>>(attn_s, temp, attn_b);

    // 10. out = attn(48x48) x v(48x4096), batch 64 -> bf16
    gemm_bf16_wmma<false, true><<<dim3(N1/64, 1, Bn*8), dim3(128), 0, stream>>>(
        attn_b, vb, ob, 48, N1, 48, 48, N1, N1,
        (long long)48*48, (long long)48*N1, (long long)48*N1);

    // 11. final = Wproj(384x384) x out(384x4096), batch 8 -> f32 d_out
    gemm_bf16_wmma<false, false><<<dim3(N1/64, Cc/64, Bn), dim3(128), 0, stream>>>(
        wprojb, ob, d_out, Cc, N1, Cc, Cc, N1, N1,
        0, (long long)Cc*N1, (long long)Cc*N1);
}
